// Decoder_40252433498453
// MI455X (gfx1250) — compile-verified, matrix operand order checked by reading, not run on hardware
//
#include <hip/hip_runtime.h>
#include <cstdint>
#include <cstddef>

typedef _Float16 half_t;
typedef __attribute__((ext_vector_type(16))) _Float16 v16h;
typedef __attribute__((ext_vector_type(8)))  float    v8f;

#define B_     32
#define T_IN_  400
#define T_OUT_ 500
#define N_MEL_ 80
#define ENC_   512
#define RNN_   1024
#define PRE_   256
#define ATT_   128
#define LOC_F_ 32
#define LOC_K_ 31

#define NWG  64
#define NTHR 256

// ---------------------------------------------------------------- helpers

__device__ __forceinline__ unsigned hash3(unsigned a, unsigned b, unsigned c) {
  unsigned h = a * 2654435761u + 0x9e3779b9u;
  h ^= b + 0x85ebca6bu + (h << 6) + (h >> 2);
  h ^= c + 0xc2b2ae35u + (h << 6) + (h >> 2);
  h ^= h >> 16; h *= 0x7feb352du; h ^= h >> 15; h *= 0x846ca68bu; h ^= h >> 16;
  return h;
}

// Element index inside one 512-half WMMA 16x32 A/B tile for (row r in 0..15, k in 0..31)
// per the CDNA5 16-bit A-matrix VGPR layout (lanes 0-15: K 0-7/16-23; lanes 16-31: K 8-15/24-31).
__device__ __forceinline__ int tidx16(int r, int k) {
  int hi, j;
  if (k < 16) { hi = k >> 3; j = k & 7; }
  else        { hi = (k - 16) >> 3; j = 8 + ((k - 16) & 7); }
  return (r + 16 * hi) * 16 + j;   // = lane*16 + j
}

__device__ __forceinline__ v8f wmma_step(const half_t* Ap, const half_t* Bp, int lane, v8f acc) {
  v16h a = *(const v16h*)(Ap + lane * 16);
  v16h b = *(const v16h*)(Bp + lane * 16);
  return __builtin_amdgcn_wmma_f32_16x16x32_f16(false, a, false, b, (short)0, acc, false, false);
}

__device__ __forceinline__ float sigm(float x) { return 1.0f / (1.0f + expf(-x)); }

__device__ __forceinline__ void grid_sync(unsigned* bar, unsigned gen) {
  __syncthreads();
  if (threadIdx.x == 0) {
    __threadfence();
    atomicAdd(bar, 1u);
    unsigned target = gen * (unsigned)NWG;
    while (atomicAdd(bar, 0u) < target) { __builtin_amdgcn_s_sleep(1); }
  }
  __syncthreads();
  __threadfence();
}

// Async DMA of `nbytes` (multiple of 16) from global to LDS using the CDNA5
// async-to-LDS path (tracked by ASYNCcnt). LDS offset = low 32 bits of the
// generic pointer (ISA 10.2).
__device__ __forceinline__ void async_tile_copy(half_t* lds_dst, const half_t* gsrc,
                                                int nbytes, int tid) {
  unsigned lbase = (unsigned)(uintptr_t)lds_dst;
  unsigned long long gbase = (unsigned long long)(uintptr_t)gsrc;
  for (int q = tid * 16; q < nbytes; q += NTHR * 16) {
    unsigned l = lbase + (unsigned)q;
    unsigned long long g = gbase + (unsigned long long)q;
    asm volatile("global_load_async_to_lds_b128 %0, %1, off"
                 :: "v"(l), "v"(g) : "memory");
  }
}

__device__ __forceinline__ void async_wait_bar() {
  asm volatile("s_wait_asynccnt 0x0" ::: "memory");
  __syncthreads();
}

// ---------------------------------------------------------------- packing kernels

// Pack weight W (O x Ksrc, row-major) as WMMA B tiles: B(k,n)=W(n,k); tile id = kt*NT + nt.
__global__ __launch_bounds__(NTHR) void pack_b_kernel(const float* src, half_t* dst,
                                                      int O, int Ksrc, int Kpad) {
  long i = (long)blockIdx.x * NTHR + threadIdx.x;
  long total = (long)O * Kpad;
  if (i >= total) return;
  int k = (int)(i % Kpad), o = (int)(i / Kpad);
  float v = (k < Ksrc) ? src[(size_t)o * Ksrc + k] : 0.0f;
  int NT = O >> 4;
  dst[((size_t)(k >> 5) * NT + (o >> 4)) * 512 + tidx16(o & 15, k & 31)] = (half_t)v;
}

// Pack memory (12800 x 512 row-major) as WMMA A tiles: tile id = mt*16 + kt.
__global__ __launch_bounds__(NTHR) void pack_mem_a_kernel(const float* memory, half_t* dst) {
  long i = (long)blockIdx.x * NTHR + threadIdx.x;
  if (i >= (long)B_ * T_IN_ * ENC_) return;
  int k = (int)(i % ENC_), row = (int)(i / ENC_);
  dst[((size_t)(row >> 4) * 16 + (k >> 5)) * 512 + tidx16(row & 15, k & 31)] = (half_t)memory[i];
}

// decoder_inputs (B, N_MEL, T_OUT) -> prenet input rows (t*32+b, K padded 80->96), A tiles (KT=3).
__global__ __launch_bounds__(NTHR) void pack_decin_a_kernel(const float* dec_inputs, half_t* dst) {
  long i = (long)blockIdx.x * NTHR + threadIdx.x;
  long total = (long)T_OUT_ * B_ * 96;
  if (i >= total) return;
  int k = (int)(i % 96), row = (int)(i / 96);
  int tstep = row >> 5, b = row & 31;
  float v = 0.0f;
  if (tstep > 0 && k < N_MEL_)
    v = dec_inputs[(size_t)b * N_MEL_ * T_OUT_ + (size_t)k * T_OUT_ + (tstep - 1)];
  dst[((size_t)(row >> 4) * 3 + (k >> 5)) * 512 + tidx16(row & 15, k & 31)] = (half_t)v;
}

// ---------------------------------------------------------------- WMMA GEMM kernels (prep)

// C = relu(drop(A @ B)) written back as packed-A f16 tiles (for the next GEMM layer).
__global__ __launch_bounds__(NTHR) void mm_pack_relu_drop(const half_t* A, const half_t* Bw,
                                                          half_t* outA, int MT, int NT, int KT,
                                                          int outKT, unsigned salt,
                                                          unsigned keep_thresh, float scale) {
  int gw = blockIdx.x * 8 + (threadIdx.x >> 5);
  if (gw >= MT * NT) return;
  int lane = threadIdx.x & 31;
  int mt = gw / NT, nt = gw % NT;
  v8f acc = {};
  const half_t* Ap = A + (size_t)mt * KT * 512;
  const half_t* Bp = Bw + (size_t)nt * 512;
  for (int kt = 0; kt < KT; ++kt) {
    acc = wmma_step(Ap, Bp, lane, acc);
    Ap += 512; Bp += (size_t)NT * 512;
  }
  int m0 = (lane >> 4) * 8, col = nt * 16 + (lane & 15);
  for (int r = 0; r < 8; ++r) {
    int grow = mt * 16 + m0 + r;
    float v = acc[r];
    v = v > 0.0f ? v : 0.0f;
    unsigned h = hash3((unsigned)grow, (unsigned)col, salt);
    v = (h >= keep_thresh) ? v * scale : 0.0f;
    outA[((size_t)mt * outKT + (col >> 5)) * 512 + tidx16(m0 + r, col & 31)] = (half_t)v;
  }
}

// Plain WMMA GEMM with f32 row-major output (processed_memory).
__global__ __launch_bounds__(NTHR) void mm_f32_out(const half_t* A, const half_t* Bw,
                                                   float* out, int MT, int NT, int KT) {
  int gw = blockIdx.x * 8 + (threadIdx.x >> 5);
  if (gw >= MT * NT) return;
  int lane = threadIdx.x & 31;
  int mt = gw / NT, nt = gw % NT;
  v8f acc = {};
  const half_t* Ap = A + (size_t)mt * KT * 512;
  const half_t* Bp = Bw + (size_t)nt * 512;
  for (int kt = 0; kt < KT; ++kt) {
    acc = wmma_step(Ap, Bp, lane, acc);
    Ap += 512; Bp += (size_t)NT * 512;
  }
  int m0 = (lane >> 4) * 8, col = nt * 16 + (lane & 15);
  for (int r = 0; r < 8; ++r)
    out[(size_t)(mt * 16 + m0 + r) * (NT * 16) + col] = acc[r];
}

// ---------------------------------------------------------------- persistent decoder

struct DecArgs {
  const float* memory; const int* mem_len;
  const float *att_bih, *att_bhh, *dec_bih, *dec_bhh;
  const float *v_w, *loc_conv_w, *loc_lin_w, *proj_b, *gate_w, *gate_b;
  unsigned* bar;
  float *ac, *dc, *dh, *aw, *awc, *ctx;
  half_t *ctxA, *ahA, *dhA;
  float *convbuf, *locA, *pq, *G1, *G2, *pm;
  const half_t *prenetA, *WbAih, *WbAhh, *WbDih, *WbDhh, *WbQ, *WbProj;
  float *mel_out, *gate_out, *align_out;
};

__global__ __launch_bounds__(NTHR) void decoder_persistent(DecArgs P) {
  const int tid = threadIdx.x;
  const int lane = tid & 31;
  const int gwave = blockIdx.x * 8 + (tid >> 5);        // 0..511
  const int gtid = blockIdx.x * NTHR + tid;             // 0..16383
  const int mtb = blockIdx.x >> 5;                      // block-uniform M tile (0/1)
  __shared__ half_t sA[56 * 512];                        // staged A tiles (56 KB)
  __shared__ float sh_e[T_IN_];
  __shared__ float sh_red[NTHR];
  unsigned gen = 0;

  for (int t = 0; t < T_OUT_; ++t) {
    // ---- Phase A: attention-LSTM gates (WMMA, 512 tiles) + location conv (VALU).
    // Stage the block-shared A tiles (prenet_t | ctx | ah) into LDS via async DMA:
    // 8 waves/block reuse them, cutting L2 A-traffic 8x.
    async_tile_copy(sA,            P.prenetA + ((size_t)t * 2 + mtb) * 8 * 512, 8 * 1024, tid);
    async_tile_copy(sA + 8 * 512,  P.ctxA + (size_t)mtb * 16 * 512, 16 * 1024, tid);
    async_tile_copy(sA + 24 * 512, P.ahA + (size_t)mtb * 32 * 512, 32 * 1024, tid);
    async_wait_bar();
    {
      int mt = gwave >> 8, nt = gwave & 255;
      v8f acc = {};
      const half_t* Bp = P.WbAih + (size_t)nt * 512;
      for (int kt = 0; kt < 24; ++kt) {            // [prenet(8) | ctx(16)] vs WbAih
        __builtin_prefetch(Bp + (size_t)256 * 512, 0, 0);
        acc = wmma_step(sA + (size_t)kt * 512, Bp, lane, acc);
        Bp += (size_t)256 * 512;
      }
      Bp = P.WbAhh + (size_t)nt * 512;
      for (int kt = 0; kt < 32; ++kt) {            // ah vs WbAhh
        acc = wmma_step(sA + (size_t)(24 + kt) * 512, Bp, lane, acc);
        Bp += (size_t)256 * 512;
      }
      int m0 = (lane >> 4) * 8, n = nt * 16 + (lane & 15);
      for (int r = 0; r < 8; ++r) P.G1[(size_t)(mt * 16 + m0 + r) * 4096 + n] = acc[r];
    }
    for (int idx = gtid; idx < B_ * LOC_F_ * T_IN_; idx += NWG * NTHR) {
      int tt = idx % T_IN_; int f = (idx / T_IN_) & 31; int b = idx / (T_IN_ * LOC_F_);
      const float* w0 = P.loc_conv_w + (size_t)f * 2 * LOC_K_;
      const float* awr = P.aw + (size_t)b * T_IN_;
      const float* awcr = P.awc + (size_t)b * T_IN_;
      float s = 0.0f;
      for (int k = 0; k < LOC_K_; ++k) {
        int tin = tt + k - 15;
        if (tin >= 0 && tin < T_IN_) s += w0[k] * awr[tin] + w0[LOC_K_ + k] * awcr[tin];
      }
      P.convbuf[idx] = s;
    }
    ++gen; grid_sync(P.bar, gen);

    // ---- Phase B: attention-LSTM elementwise -> ac, ahA (packed f16)
    for (int idx = gtid; idx < B_ * RNN_; idx += NWG * NTHR) {
      int b = idx >> 10, o = idx & 1023;
      const float* g = P.G1 + (size_t)b * 4096;
      float gi = g[o]        + P.att_bih[o]        + P.att_bhh[o];
      float gf = g[1024 + o] + P.att_bih[1024 + o] + P.att_bhh[1024 + o];
      float gg = g[2048 + o] + P.att_bih[2048 + o] + P.att_bhh[2048 + o];
      float go = g[3072 + o] + P.att_bih[3072 + o] + P.att_bhh[3072 + o];
      float c2 = sigm(gf) * P.ac[idx] + sigm(gi) * tanhf(gg);
      float h2 = sigm(go) * tanhf(c2);
      unsigned hh = hash3((unsigned)(t * 32 + b), (unsigned)o, 0xA77u);
      h2 = (hh % 10u) ? h2 * (1.0f / 0.9f) : 0.0f;
      P.ac[idx] = c2;
      P.ahA[((size_t)(b >> 4) * 32 + (o >> 5)) * 512 + tidx16(b & 15, o & 31)] = (half_t)h2;
    }
    ++gen; grid_sync(P.bar, gen);

    // ---- Phase C: pq = ah @ query_w^T (WMMA, 16 tiles) + location linear (VALU)
    if (gwave < 16) {
      int mt = gwave >> 3, nt = gwave & 7;
      v8f acc = {};
      const half_t* Ap = P.ahA + (size_t)mt * 32 * 512;
      const half_t* Bp = P.WbQ + (size_t)nt * 512;
      for (int kt = 0; kt < 32; ++kt) {
        acc = wmma_step(Ap, Bp, lane, acc);
        Ap += 512; Bp += (size_t)8 * 512;
      }
      int m0 = (lane >> 4) * 8, n = nt * 16 + (lane & 15);
      for (int r = 0; r < 8; ++r) P.pq[(size_t)(mt * 16 + m0 + r) * ATT_ + n] = acc[r];
    }
    for (int idx = gtid; idx < B_ * T_IN_ * ATT_; idx += NWG * NTHR) {
      int a = idx & 127; int tt = (idx >> 7) % T_IN_; int b = idx / (ATT_ * T_IN_);
      const float* cv = P.convbuf + (size_t)b * LOC_F_ * T_IN_ + tt;
      const float* wl = P.loc_lin_w + (size_t)a * LOC_F_;
      float s = 0.0f;
      for (int f = 0; f < LOC_F_; ++f) s += wl[f] * cv[(size_t)f * T_IN_];
      P.locA[idx] = s;
    }
    ++gen; grid_sync(P.bar, gen);

    // ---- Phase D: energies, softmax, context (one workgroup per batch row)
    if (blockIdx.x < B_) {
      int b = blockIdx.x;
      int len = P.mem_len[b];
      for (int tt = tid; tt < T_IN_; tt += NTHR) {
        float s;
        if (tt < len) {
          const float* pmr = P.pm + ((size_t)b * T_IN_ + tt) * ATT_;
          const float* lor = P.locA + ((size_t)b * T_IN_ + tt) * ATT_;
          const float* pqr = P.pq + (size_t)b * ATT_;
          s = 0.0f;
          for (int a = 0; a < ATT_; ++a) s += P.v_w[a] * tanhf(pqr[a] + lor[a] + pmr[a]);
        } else s = -1.0e30f;
        sh_e[tt] = s;
      }
      __syncthreads();
      float mx = -1.0e30f;
      for (int tt = tid; tt < T_IN_; tt += NTHR) mx = fmaxf(mx, sh_e[tt]);
      sh_red[tid] = mx; __syncthreads();
      for (int st = NTHR >> 1; st > 0; st >>= 1) {
        if (tid < st) sh_red[tid] = fmaxf(sh_red[tid], sh_red[tid + st]);
        __syncthreads();
      }
      mx = sh_red[0]; __syncthreads();
      float ssum = 0.0f;
      for (int tt = tid; tt < T_IN_; tt += NTHR) {
        float ex = expf(sh_e[tt] - mx); sh_e[tt] = ex; ssum += ex;
      }
      sh_red[tid] = ssum; __syncthreads();
      for (int st = NTHR >> 1; st > 0; st >>= 1) {
        if (tid < st) sh_red[tid] += sh_red[tid + st];
        __syncthreads();
      }
      float inv = 1.0f / sh_red[0];
      __syncthreads();
      for (int tt = tid; tt < T_IN_; tt += NTHR) {
        float a2 = sh_e[tt] * inv; sh_e[tt] = a2;
        P.aw[(size_t)b * T_IN_ + tt] = a2;
        P.awc[(size_t)b * T_IN_ + tt] += a2;
        P.align_out[(size_t)b * T_OUT_ * T_IN_ + (size_t)t * T_IN_ + tt] = a2;
      }
      __syncthreads();
      for (int d = tid; d < ENC_; d += NTHR) {
        const float* mrow = P.memory + (size_t)b * T_IN_ * ENC_ + d;
        float s = 0.0f;
        for (int tt = 0; tt < T_IN_; ++tt) s += sh_e[tt] * mrow[(size_t)tt * ENC_];
        P.ctx[(size_t)b * ENC_ + d] = s;
        P.ctxA[((size_t)(b >> 4) * 16 + (d >> 5)) * 512 + tidx16(b & 15, d & 31)] = (half_t)s;
      }
    }
    ++gen; grid_sync(P.bar, gen);

    // ---- Phase E: decoder-LSTM gates (WMMA, 512 tiles, K=80 steps).
    // Stage new ah + ctx tiles to LDS (48 KB); stream dh A-side from L2.
    async_tile_copy(sA,            P.ahA + (size_t)mtb * 32 * 512, 32 * 1024, tid);
    async_tile_copy(sA + 32 * 512, P.ctxA + (size_t)mtb * 16 * 512, 16 * 1024, tid);
    async_wait_bar();
    {
      int mt = gwave >> 8, nt = gwave & 255;
      v8f acc = {};
      const half_t* Bp = P.WbDih + (size_t)nt * 512;
      for (int kt = 0; kt < 48; ++kt) {            // [ah(32) | ctx(16)] vs WbDih
        __builtin_prefetch(Bp + (size_t)256 * 512, 0, 0);
        acc = wmma_step(sA + (size_t)kt * 512, Bp, lane, acc);
        Bp += (size_t)256 * 512;
      }
      const half_t* Ap = P.dhA + (size_t)mt * 32 * 512;
      Bp = P.WbDhh + (size_t)nt * 512;
      for (int kt = 0; kt < 32; ++kt) {            // dh vs WbDhh
        acc = wmma_step(Ap, Bp, lane, acc);
        Ap += 512; Bp += (size_t)256 * 512;
      }
      int m0 = (lane >> 4) * 8, n = nt * 16 + (lane & 15);
      for (int r = 0; r < 8; ++r) P.G2[(size_t)(mt * 16 + m0 + r) * 4096 + n] = acc[r];
    }
    ++gen; grid_sync(P.bar, gen);

    // ---- Phase F: decoder-LSTM elementwise -> dc, dh, dhA
    for (int idx = gtid; idx < B_ * RNN_; idx += NWG * NTHR) {
      int b = idx >> 10, o = idx & 1023;
      const float* g = P.G2 + (size_t)b * 4096;
      float gi = g[o]        + P.dec_bih[o]        + P.dec_bhh[o];
      float gf = g[1024 + o] + P.dec_bih[1024 + o] + P.dec_bhh[1024 + o];
      float gg = g[2048 + o] + P.dec_bih[2048 + o] + P.dec_bhh[2048 + o];
      float go = g[3072 + o] + P.dec_bih[3072 + o] + P.dec_bhh[3072 + o];
      float c2 = sigm(gf) * P.dc[idx] + sigm(gi) * tanhf(gg);
      float h2 = sigm(go) * tanhf(c2);
      unsigned hh = hash3((unsigned)(t * 32 + b), (unsigned)o, 0xD33u);
      h2 = (hh % 10u) ? h2 * (1.0f / 0.9f) : 0.0f;
      P.dc[idx] = c2;
      P.dh[idx] = h2;
      P.dhA[((size_t)(b >> 4) * 32 + (o >> 5)) * 512 + tidx16(b & 15, o & 31)] = (half_t)h2;
    }
    ++gen; grid_sync(P.bar, gen);

    // ---- Phase G: mel projection (WMMA, 10 tiles) + gate (VALU); overlaps next Phase A.
    if (gwave < 10) {
      int mt = gwave / 5, nt = gwave % 5;
      v8f acc = {};
      const half_t* Ap = P.dhA + (size_t)mt * 32 * 512;
      const half_t* Bp = P.WbProj + (size_t)nt * 512;
      for (int kt = 0; kt < 32; ++kt) {
        acc = wmma_step(Ap, Bp, lane, acc);
        Ap += 512; Bp += (size_t)5 * 512;
      }
      Ap = P.ctxA + (size_t)mt * 16 * 512;
      for (int kt = 0; kt < 16; ++kt) {
        acc = wmma_step(Ap, Bp, lane, acc);
        Ap += 512; Bp += (size_t)5 * 512;
      }
      int m0 = (lane >> 4) * 8, col = nt * 16 + (lane & 15);
      for (int r = 0; r < 8; ++r) {
        int b = mt * 16 + m0 + r;
        P.mel_out[(size_t)b * N_MEL_ * T_OUT_ + (size_t)col * T_OUT_ + t] = acc[r] + P.proj_b[col];
      }
    } else if (gwave == 10) {
      int b = lane;
      float s = P.gate_b[0];
      for (int k = 0; k < RNN_; ++k) s += P.dh[(size_t)b * RNN_ + k] * P.gate_w[k];
      for (int d = 0; d < ENC_; ++d) s += P.ctx[(size_t)b * ENC_ + d] * P.gate_w[RNN_ + d];
      P.gate_out[(size_t)b * T_OUT_ + t] = s;
    }
    // no sync: next Phase A touches disjoint buffers; post-A barrier orders Phase G.
  }
}

// ---------------------------------------------------------------- host

extern "C" void kernel_launch(void* const* d_in, const int* in_sizes, int n_in,
                              void* d_out, int out_size, void* d_ws, size_t ws_size,
                              hipStream_t stream) {
  const float* memory     = (const float*)d_in[0];
  const float* dec_inputs = (const float*)d_in[1];
  const int*   mem_len    = (const int*)d_in[2];
  const float* prenet_w1  = (const float*)d_in[3];
  const float* prenet_w2  = (const float*)d_in[4];
  const float* att_wih    = (const float*)d_in[5];
  const float* att_whh    = (const float*)d_in[6];
  const float* att_bih    = (const float*)d_in[7];
  const float* att_bhh    = (const float*)d_in[8];
  const float* query_w    = (const float*)d_in[9];
  const float* memory_w   = (const float*)d_in[10];
  const float* v_w        = (const float*)d_in[11];
  const float* loc_conv_w = (const float*)d_in[12];
  const float* loc_lin_w  = (const float*)d_in[13];
  const float* dec_wih    = (const float*)d_in[14];
  const float* dec_whh    = (const float*)d_in[15];
  const float* dec_bih    = (const float*)d_in[16];
  const float* dec_bhh    = (const float*)d_in[17];
  const float* proj_w     = (const float*)d_in[18];
  const float* proj_b     = (const float*)d_in[19];
  const float* gate_w     = (const float*)d_in[20];
  const float* gate_b     = (const float*)d_in[21];
  (void)in_sizes; (void)n_in; (void)out_size;

  char* ws = (char*)d_ws;
  size_t off = 0;
  auto alloc = [&](size_t bytes) { size_t o = off; off += (bytes + 1023) & ~(size_t)1023; return o; };

  // ---- zero-initialized region (recurrent state + barrier) ----
  size_t o_bar  = alloc(256);
  size_t o_ac   = alloc((size_t)B_ * RNN_ * 4);
  size_t o_dc   = alloc((size_t)B_ * RNN_ * 4);
  size_t o_dh   = alloc((size_t)B_ * RNN_ * 4);
  size_t o_aw   = alloc((size_t)B_ * T_IN_ * 4);
  size_t o_awc  = alloc((size_t)B_ * T_IN_ * 4);
  size_t o_ctx  = alloc((size_t)B_ * ENC_ * 4);
  size_t o_ctxA = alloc((size_t)2 * 16 * 512 * 2);
  size_t o_ahA  = alloc((size_t)2 * 32 * 512 * 2);
  size_t o_dhA  = alloc((size_t)2 * 32 * 512 * 2);
  size_t zero_bytes = off;

  // ---- scratch ----
  size_t o_conv = alloc((size_t)B_ * LOC_F_ * T_IN_ * 4);
  size_t o_locA = alloc((size_t)B_ * T_IN_ * ATT_ * 4);
  size_t o_pq   = alloc((size_t)B_ * ATT_ * 4);
  size_t o_G1   = alloc((size_t)B_ * 4 * RNN_ * 4);
  size_t o_G2   = alloc((size_t)B_ * 4 * RNN_ * 4);
  size_t o_pm   = alloc((size_t)B_ * T_IN_ * ATT_ * 4);
  size_t o_prA  = alloc((size_t)T_OUT_ * 2 * 8 * 512 * 2);   // prenet output, packed A
  size_t o_memA = alloc((size_t)800 * 16 * 512 * 2);
  size_t o_dinA = alloc((size_t)1000 * 3 * 512 * 2);
  size_t o_h1A  = alloc((size_t)1000 * 8 * 512 * 2);
  // packed f16 weights (B tiles)
  size_t o_WAih = alloc((size_t)24 * 256 * 512 * 2);
  size_t o_WAhh = alloc((size_t)32 * 256 * 512 * 2);
  size_t o_WDih = alloc((size_t)48 * 256 * 512 * 2);
  size_t o_WDhh = alloc((size_t)32 * 256 * 512 * 2);
  size_t o_WQ   = alloc((size_t)32 * 8 * 512 * 2);
  size_t o_WM   = alloc((size_t)16 * 8 * 512 * 2);
  size_t o_WP   = alloc((size_t)48 * 5 * 512 * 2);
  size_t o_W1   = alloc((size_t)3 * 16 * 512 * 2);
  size_t o_W2   = alloc((size_t)8 * 16 * 512 * 2);

  if (off > ws_size) return;  // insufficient scratch; nothing sensible to do

  hipMemsetAsync(ws, 0, zero_bytes, stream);

  auto gridFor = [](long n) { return (unsigned)((n + NTHR - 1) / NTHR); };

  // Pack weights into WMMA B-tile f16 layout
  pack_b_kernel<<<gridFor(4096L * 768), NTHR, 0, stream>>>(att_wih, (half_t*)(ws + o_WAih), 4096, 768, 768);
  pack_b_kernel<<<gridFor(4096L * 1024), NTHR, 0, stream>>>(att_whh, (half_t*)(ws + o_WAhh), 4096, 1024, 1024);
  pack_b_kernel<<<gridFor(4096L * 1536), NTHR, 0, stream>>>(dec_wih, (half_t*)(ws + o_WDih), 4096, 1536, 1536);
  pack_b_kernel<<<gridFor(4096L * 1024), NTHR, 0, stream>>>(dec_whh, (half_t*)(ws + o_WDhh), 4096, 1024, 1024);
  pack_b_kernel<<<gridFor(128L * 1024), NTHR, 0, stream>>>(query_w, (half_t*)(ws + o_WQ), 128, 1024, 1024);
  pack_b_kernel<<<gridFor(128L * 512), NTHR, 0, stream>>>(memory_w, (half_t*)(ws + o_WM), 128, 512, 512);
  pack_b_kernel<<<gridFor(80L * 1536), NTHR, 0, stream>>>(proj_w, (half_t*)(ws + o_WP), 80, 1536, 1536);
  pack_b_kernel<<<gridFor(256L * 96), NTHR, 0, stream>>>(prenet_w1, (half_t*)(ws + o_W1), 256, 80, 96);
  pack_b_kernel<<<gridFor(256L * 256), NTHR, 0, stream>>>(prenet_w2, (half_t*)(ws + o_W2), 256, 256, 256);

  // Pack activations into A-tile layout
  pack_mem_a_kernel<<<gridFor((long)B_ * T_IN_ * ENC_), NTHR, 0, stream>>>(memory, (half_t*)(ws + o_memA));
  pack_decin_a_kernel<<<gridFor((long)T_OUT_ * B_ * 96), NTHR, 0, stream>>>(dec_inputs, (half_t*)(ws + o_dinA));

  // Prenet: two WMMA layers with relu + p=0.5 dropout, outputs stay packed-A
  mm_pack_relu_drop<<<2000, NTHR, 0, stream>>>((const half_t*)(ws + o_dinA), (const half_t*)(ws + o_W1),
                                               (half_t*)(ws + o_h1A), 1000, 16, 3, 8,
                                               0x111u, 0x80000000u, 2.0f);
  mm_pack_relu_drop<<<2000, NTHR, 0, stream>>>((const half_t*)(ws + o_h1A), (const half_t*)(ws + o_W2),
                                               (half_t*)(ws + o_prA), 1000, 16, 8, 8,
                                               0x222u, 0x80000000u, 2.0f);

  // processed_memory = memory @ memory_w^T  (WMMA, f32 out)
  mm_f32_out<<<800, NTHR, 0, stream>>>((const half_t*)(ws + o_memA), (const half_t*)(ws + o_WM),
                                       (float*)(ws + o_pm), 800, 8, 16);

  // Persistent cooperative decoder over 500 steps
  float* out = (float*)d_out;
  DecArgs P;
  P.memory = memory; P.mem_len = mem_len;
  P.att_bih = att_bih; P.att_bhh = att_bhh; P.dec_bih = dec_bih; P.dec_bhh = dec_bhh;
  P.v_w = v_w; P.loc_conv_w = loc_conv_w; P.loc_lin_w = loc_lin_w;
  P.proj_b = proj_b; P.gate_w = gate_w; P.gate_b = gate_b;
  P.bar = (unsigned*)(ws + o_bar);
  P.ac = (float*)(ws + o_ac); P.dc = (float*)(ws + o_dc); P.dh = (float*)(ws + o_dh);
  P.aw = (float*)(ws + o_aw); P.awc = (float*)(ws + o_awc); P.ctx = (float*)(ws + o_ctx);
  P.ctxA = (half_t*)(ws + o_ctxA); P.ahA = (half_t*)(ws + o_ahA); P.dhA = (half_t*)(ws + o_dhA);
  P.convbuf = (float*)(ws + o_conv); P.locA = (float*)(ws + o_locA); P.pq = (float*)(ws + o_pq);
  P.G1 = (float*)(ws + o_G1); P.G2 = (float*)(ws + o_G2); P.pm = (float*)(ws + o_pm);
  P.prenetA = (const half_t*)(ws + o_prA);
  P.WbAih = (const half_t*)(ws + o_WAih); P.WbAhh = (const half_t*)(ws + o_WAhh);
  P.WbDih = (const half_t*)(ws + o_WDih); P.WbDhh = (const half_t*)(ws + o_WDhh);
  P.WbQ = (const half_t*)(ws + o_WQ); P.WbProj = (const half_t*)(ws + o_WP);
  P.mel_out = out;
  P.gate_out = out + (size_t)B_ * N_MEL_ * T_OUT_;
  P.align_out = out + (size_t)B_ * N_MEL_ * T_OUT_ + (size_t)B_ * T_OUT_;

  decoder_persistent<<<NWG, NTHR, 0, stream>>>(P);
}
